// GRUClassifier_6837587936192
// MI455X (gfx1250) — matrix operand order repaired and checked by reading, not compiled, not run on hardware
//
#include <hip/hip_runtime.h>

// ---------------------------------------------------------------------------
// GRU classifier for MI455X (gfx1250, wave32, WMMA bf16 16x16x32, f32 accum)
// ---------------------------------------------------------------------------

typedef __bf16 bf16;
typedef __attribute__((ext_vector_type(8)))  bf16  v8bf;
typedef __attribute__((ext_vector_type(16))) bf16  v16bf;
typedef __attribute__((ext_vector_type(8)))  float v8f;

union V16 { v16bf v; v8bf h[2]; };

static constexpr int Bn   = 64;
static constexpr int Tn   = 256;
static constexpr int En   = 300;
static constexpr int EPAD = 320;
static constexpr int Hn   = 512;
static constexpr int Gn   = 1536;          // 3*H  (r, z, n)
static constexpr int NROW = Bn * Tn;       // 16384

// workspace layout (bytes), all 256-aligned
static constexpr size_t OFF_X0     = 0;                         // [NROW][320] bf16
static constexpr size_t OFF_WIH0   = 10485760;                  // [2][G][320] bf16
static constexpr size_t OFF_WHH0   = 12451840;                  // [2][G][512] bf16
static constexpr size_t OFF_WIH1   = 15597568;                  // [2][G][1024] bf16
static constexpr size_t OFF_WHH1   = 21889024;                  // [2][G][512] bf16
static constexpr size_t OFF_XP     = 25034752;                  // [2][NROW][G] bf16
static constexpr size_t OFF_H0     = 125698048;                 // [NROW][1024] bf16
static constexpr size_t OFF_POOLED = 159252480;                 // [64][1024] f32

__device__ inline bf16 f2bf(float f) {
  unsigned u = __builtin_bit_cast(unsigned, f);
  u += 0x7FFFu + ((u >> 16) & 1u);
  unsigned short s = (unsigned short)(u >> 16);
  return __builtin_bit_cast(bf16, s);
}
__device__ inline float bf2f(bf16 b) {
  unsigned u = ((unsigned)__builtin_bit_cast(unsigned short, b)) << 16;
  return __builtin_bit_cast(float, u);
}
__device__ inline float sigm(float x)      { return 1.f / (1.f + __expf(-x)); }
__device__ inline float tanh_fast(float x) { return 2.f / (1.f + __expf(-2.f * x)) - 1.f; }

// -------------------------- weight prep kernels ----------------------------

__global__ void k_f32_to_bf16(const float* __restrict__ src, bf16* __restrict__ dst, int n) {
  int i = blockIdx.x * blockDim.x + threadIdx.x;
  if (i < n) dst[i] = f2bf(src[i]);
}

// [2*G][300] f32 -> [2*G][320] bf16, zero-padded
__global__ void k_pad_wih0(const float* __restrict__ src, bf16* __restrict__ dst) {
  int row = blockIdx.x;
  for (int c = threadIdx.x; c < EPAD; c += blockDim.x) {
    float v = (c < En) ? src[(size_t)row * En + c] : 0.f;
    dst[(size_t)row * EPAD + c] = f2bf(v);
  }
}

// gather embeddings: X0[m][0:320] bf16 (zero pad cols 300..319)
__global__ void k_embed(const int* __restrict__ x, const float* __restrict__ emb,
                        bf16* __restrict__ X0) {
  int m = blockIdx.x;
  int idx = x[m];
  for (int c = threadIdx.x; c < EPAD; c += blockDim.x) {
    float v = (c < En) ? emb[(size_t)idx * En + c] : 0.f;
    X0[(size_t)m * EPAD + c] = f2bf(v);
  }
}

// ------------------- batched input projection GEMM -------------------------
// XP[dir][m][g] = sum_k X[m][k] * W[dir][g][k] + b[dir][g]   (stored bf16)
// block = 256 thr (8 waves); wave -> 16 rows x 64 cols (4 WMMA tiles)
__global__ __launch_bounds__(256)
void k_xproj(const bf16* __restrict__ X, const bf16* __restrict__ W,
             const float* __restrict__ bias, bf16* __restrict__ XP, int Kpad) {
  const int lane = threadIdx.x & 31;
  const int wv   = threadIdx.x >> 5;
  const int lo   = lane & 15;
  const int hi   = lane >> 4;
  const int dir  = blockIdx.z;
  const int m0   = blockIdx.x * 128 + wv * 16;
  const int g0   = blockIdx.y * 64;

  const bf16* Wd = W + (size_t)dir * Gn * Kpad;

  // A: 16x32 bf16 tile. lane: row = lo; K halves at base,{+8}, base = hi?16:0
  const bf16* Aptr = X + (size_t)(m0 + lo) * Kpad + (hi ? 16 : 0);
  // B: 32x16 tile. lane: col N = lo, 16 contiguous K halves at 16*hi
  const int koff = 16 * hi;

  v8f acc0 = {}, acc1 = {}, acc2 = {}, acc3 = {};
  for (int k0 = 0; k0 < Kpad; k0 += 32) {
    V16 a;
    a.h[0] = *(const v8bf*)(Aptr + k0);
    a.h[1] = *(const v8bf*)(Aptr + k0 + 8);
    const bf16* bp0 = Wd + (size_t)(g0 +  0 + lo) * Kpad + k0 + koff;
    const bf16* bp1 = Wd + (size_t)(g0 + 16 + lo) * Kpad + k0 + koff;
    const bf16* bp2 = Wd + (size_t)(g0 + 32 + lo) * Kpad + k0 + koff;
    const bf16* bp3 = Wd + (size_t)(g0 + 48 + lo) * Kpad + k0 + koff;
    acc0 = __builtin_amdgcn_wmma_f32_16x16x32_bf16(false, a.v, false, *(const v16bf*)bp0, (short)0, acc0, false, false);
    acc1 = __builtin_amdgcn_wmma_f32_16x16x32_bf16(false, a.v, false, *(const v16bf*)bp1, (short)0, acc1, false, false);
    acc2 = __builtin_amdgcn_wmma_f32_16x16x32_bf16(false, a.v, false, *(const v16bf*)bp2, (short)0, acc2, false, false);
    acc3 = __builtin_amdgcn_wmma_f32_16x16x32_bf16(false, a.v, false, *(const v16bf*)bp3, (short)0, acc3, false, false);
  }

  // C/D: vgpr i -> row m0 + i + 8*hi, col g0 + 16*q + lo
  bf16* XPd = XP + (size_t)dir * NROW * Gn;
  const int crow0 = m0 + 8 * hi;
  v8f accs[4] = {acc0, acc1, acc2, acc3};
  #pragma unroll
  for (int q = 0; q < 4; ++q) {
    const int g = g0 + 16 * q + lo;
    const float bv = bias[dir * Gn + g];
    #pragma unroll
    for (int i = 0; i < 8; ++i)
      XPd[(size_t)(crow0 + i) * Gn + g] = f2bf(accs[q][i] + bv);
  }
}

// ------------------------ persistent GRU recurrence ------------------------
// grid = (B/16, 2 dirs); block = 1024 thr (32 waves).
// Wave w owns gate columns [w*16, w*16+16) of r, z, n simultaneously, so the
// gate combine is pure register math. h carry fp32 in regs. The bf16 copy of
// h (next step's WMMA A operand) lives in ping-pong LDS buffers so only ONE
// block barrier is needed per timestep: step s reads hb[s&1], writes
// hb[1-(s&1)]; the end-of-step barrier orders writes vs. next step's reads.
__global__ __launch_bounds__(1024)
void k_gru(const bf16* __restrict__ XP, const bf16* __restrict__ Whh,
           const float* __restrict__ bhh, bf16* __restrict__ Hout,
           float* __restrict__ pooled) {
  const int dir  = blockIdx.y;
  const int b0   = blockIdx.x * 16;
  const int tid  = threadIdx.x;
  const int lane = tid & 31;
  const int wv   = tid >> 5;
  const int lo   = lane & 15;
  const int hi   = lane >> 4;

  __shared__ __align__(16) bf16 hb[2][16][520];   // 33 KB, padded pitch

  for (int i = tid; i < 2 * 16 * 520; i += 1024)
    ((bf16*)hb)[i] = __builtin_bit_cast(bf16, (unsigned short)0);

  const bf16*  XPd = XP  + (size_t)dir * NROW * Gn;
  const bf16*  Whd = Whh + (size_t)dir * Gn * Hn;
  const float* bhd = bhh + dir * Gn;

  const int col = wv * 16 + lo;                 // this lane's h column (0..511)
  const float bhr = bhd[col];
  const float bhz = bhd[col + 512];
  const float bhn = bhd[col + 1024];

  const bf16* Bpr = Whd + (size_t)(col        ) * Hn + 16 * hi;
  const bf16* Bpz = Whd + (size_t)(col +  512 ) * Hn + 16 * hi;
  const bf16* Bpn = Whd + (size_t)(col + 1024 ) * Hn + 16 * hi;

  const int rbase = 8 * hi;
  const int abase = hi ? 16 : 0;

  float hprev[8], rmax[8];
  #pragma unroll
  for (int i = 0; i < 8; ++i) { hprev[i] = 0.f; rmax[i] = -3.4e38f; }

  __syncthreads();

  for (int s = 0; s < Tn; ++s) {
    const int t  = (dir == 0) ? s : (Tn - 1 - s);
    const int pr = s & 1;          // buffer holding h(t-1)
    const int pw = pr ^ 1;         // buffer receiving h(t)

    // Hoist the xp_t loads: independent of the GEMM, so the WMMA chain below
    // hides their global-memory latency.
    float xr[8], xz[8], xn[8];
    #pragma unroll
    for (int i = 0; i < 8; ++i) {
      const size_t m = (size_t)(b0 + rbase + i) * Tn + t;
      xr[i] = bf2f(XPd[m * Gn + col]);
      xz[i] = bf2f(XPd[m * Gn + col + 512]);
      xn[i] = bf2f(XPd[m * Gn + col + 1024]);
    }

    v8f ar = {}, az = {}, an = {};
    for (int k0 = 0; k0 < Hn; k0 += 32) {
      V16 a;
      a.h[0] = *(const v8bf*)&hb[pr][lo][k0 + abase];
      a.h[1] = *(const v8bf*)&hb[pr][lo][k0 + abase + 8];
      v16bf br = *(const v16bf*)(Bpr + k0);
      v16bf bz = *(const v16bf*)(Bpz + k0);
      v16bf bn = *(const v16bf*)(Bpn + k0);
      ar = __builtin_amdgcn_wmma_f32_16x16x32_bf16(false, a.v, false, br, (short)0, ar, false, false);
      az = __builtin_amdgcn_wmma_f32_16x16x32_bf16(false, a.v, false, bz, (short)0, az, false, false);
      an = __builtin_amdgcn_wmma_f32_16x16x32_bf16(false, a.v, false, bn, (short)0, an, false, false);
    }

    #pragma unroll
    for (int i = 0; i < 8; ++i) {
      const int row = rbase + i;
      const size_t m = (size_t)(b0 + row) * Tn + t;
      const float r = sigm(ar[i] + bhr + xr[i]);
      const float z = sigm(az[i] + bhz + xz[i]);
      const float n = tanh_fast(xn[i] + r * (an[i] + bhn));
      const float h = (1.f - z) * n + z * hprev[i];
      hprev[i] = h;
      hb[pw][row][col] = f2bf(h);
      if (Hout)   Hout[m * 1024 + dir * 512 + col] = f2bf(h);
      if (pooled) rmax[i] = fmaxf(rmax[i], h);
    }

    __syncthreads();   // order hb[pw] writes before next step's reads
  }

  if (pooled) {
    #pragma unroll
    for (int i = 0; i < 8; ++i)
      pooled[(size_t)(b0 + rbase + i) * 1024 + dir * 512 + col] = rmax[i];
  }
}

// ----------------------------- classifier ----------------------------------
__global__ __launch_bounds__(128)
void k_cls(const float* __restrict__ pooled, const float* __restrict__ w1,
           const float* __restrict__ b1, const float* __restrict__ w2,
           const float* __restrict__ b2, float* __restrict__ out) {
  const int b = blockIdx.x;
  const int j = threadIdx.x;
  const float* p  = pooled + (size_t)b * 1024;
  const float* wr = w1 + (size_t)j * 1024;
  float acc = b1[j];
  for (int k = 0; k < 1024; ++k) acc += p[k] * wr[k];
  __shared__ float sh[128];
  sh[j] = fmaxf(acc, 0.f);
  __syncthreads();
  if (j < 2) {
    float o = b2[j];
    for (int k = 0; k < 128; ++k) o += sh[k] * w2[j * 128 + k];
    out[b * 2 + j] = o;
  }
}

// ------------------------------- launch ------------------------------------
extern "C" void kernel_launch(void* const* d_in, const int* in_sizes, int n_in,
                              void* d_out, int out_size, void* d_ws, size_t ws_size,
                              hipStream_t stream) {
  (void)in_sizes; (void)n_in; (void)out_size; (void)ws_size;

  const int*   x     = (const int*)  d_in[0];
  const float* emb   = (const float*)d_in[1];
  const float* w_ih0 = (const float*)d_in[2];
  const float* w_hh0 = (const float*)d_in[3];
  const float* b_ih0 = (const float*)d_in[4];
  const float* b_hh0 = (const float*)d_in[5];
  const float* w_ih1 = (const float*)d_in[6];
  const float* w_hh1 = (const float*)d_in[7];
  const float* b_ih1 = (const float*)d_in[8];
  const float* b_hh1 = (const float*)d_in[9];
  const float* w1    = (const float*)d_in[10];
  const float* b1    = (const float*)d_in[11];
  const float* w2    = (const float*)d_in[12];
  const float* b2    = (const float*)d_in[13];

  char* ws = (char*)d_ws;
  bf16*  X0     = (bf16*) (ws + OFF_X0);
  bf16*  wih0b  = (bf16*) (ws + OFF_WIH0);
  bf16*  whh0b  = (bf16*) (ws + OFF_WHH0);
  bf16*  wih1b  = (bf16*) (ws + OFF_WIH1);
  bf16*  whh1b  = (bf16*) (ws + OFF_WHH1);
  bf16*  XP     = (bf16*) (ws + OFF_XP);
  bf16*  H0     = (bf16*) (ws + OFF_H0);
  float* pooled = (float*)(ws + OFF_POOLED);

  // weights -> bf16 (once per call; cheap vs. total work)
  k_pad_wih0<<<2 * Gn, 64, 0, stream>>>(w_ih0, wih0b);
  int n;
  n = 2 * Gn * Hn;   k_f32_to_bf16<<<(n + 255) / 256, 256, 0, stream>>>(w_hh0, whh0b, n);
  n = 2 * Gn * 1024; k_f32_to_bf16<<<(n + 255) / 256, 256, 0, stream>>>(w_ih1, wih1b, n);
  n = 2 * Gn * Hn;   k_f32_to_bf16<<<(n + 255) / 256, 256, 0, stream>>>(w_hh1, whh1b, n);

  // embedding gather
  k_embed<<<NROW, 64, 0, stream>>>(x, emb, X0);

  // layer 0: batched input projection, then recurrence (writes H0 sequence)
  k_xproj<<<dim3(NROW / 128, Gn / 64, 2), 256, 0, stream>>>(X0, wih0b, b_ih0, XP, EPAD);
  k_gru  <<<dim3(Bn / 16, 2), 1024, 0, stream>>>(XP, whh0b, b_hh0, H0, nullptr);

  // layer 1: projection over H0 (K=1024), recurrence with fused max-pool
  k_xproj<<<dim3(NROW / 128, Gn / 64, 2), 256, 0, stream>>>(H0, wih1b, b_ih1, XP, 1024);
  k_gru  <<<dim3(Bn / 16, 2), 1024, 0, stream>>>(XP, whh1b, b_hh1, nullptr, pooled);

  // classifier head
  k_cls<<<Bn, 128, 0, stream>>>(pooled, w1, b1, w2, b2, (float*)d_out);
}
